// CIN_51539607712
// MI455X (gfx1250) — matrix operand order repaired and checked by reading, not compiled, not run on hardware
//
#include <hip/hip_runtime.h>
#include <hip/hip_bf16.h>
#include <stdint.h>

typedef _Float16 v4h  __attribute__((ext_vector_type(4)));
typedef _Float16 v8h  __attribute__((ext_vector_type(8)));
typedef _Float16 v16h __attribute__((ext_vector_type(16)));
typedef float    v8f  __attribute__((ext_vector_type(8)));

#define BATCH 1024
#define NFEAT 32
#define EMB   16
#define HID   128

#define CH        4                    // weight slabs (k-steps) per async chunk
#define SLAB_B    (HID * NFEAT * 2)    // 8192 bytes per k-step slab
#define CHUNK_B   (CH * SLAB_B)        // 32768 bytes per chunk
// dynamic LDS layout (bytes)
#define OFF_XL    (2 * CHUNK_B)        // 65536 : xl[h][row][batch] 128*16*8 f16 = 32KB
#define OFF_XB    (2 * CHUNK_B + 32768)// 98304 : x [m][row][batch]  32*16*8 f16 =  8KB
#define LDS_BYTES (2 * CHUNK_B + 32768 + 8192)  // 106496

// ---------------- prep kernels: fp32 -> f16, layout swizzles ----------------

// X[(b*16+d)*32 + m] = (f16) inputs[b, m, d]
__global__ __launch_bounds__(256) void prep_x(const float* __restrict__ in,
                                              _Float16* __restrict__ X) {
    int i = blockIdx.x * 256 + threadIdx.x;
    if (i >= BATCH * EMB * NFEAT) return;
    int m   = i & 31;
    int pos = i >> 5;
    int b   = pos >> 4;
    int d   = pos & 15;
    X[i] = (_Float16)in[(b * NFEAT + m) * EMB + d];
}

// Wt[(h*128 + n)*32 + m] = (f16) W[(h*32+m)*128 + n]
__global__ __launch_bounds__(256) void prep_w(const float* __restrict__ W,
                                              _Float16* __restrict__ Wt, int nh) {
    int i = blockIdx.x * 256 + threadIdx.x;
    if (i >= nh * HID * NFEAT) return;
    int m = i & 31;
    int n = (i >> 5) & 127;
    int h = i >> 12;
    Wt[i] = (_Float16)W[(h * NFEAT + m) * HID + n];
}

// ---------------- main kernel ----------------
// Block: 8 waves, 8 batch elements.  Wave w: batches bgrp*4..+3 (bgrp=w>>2),
// N-groups {2*gp, 2*gp+1} (gp=w&3).  8 accumulators per wave (4 batch x 2 grp).
// Weight slabs stream Global->LDS with global_load_async_to_lds_b128, double
// buffered per 4-slab chunk; B fragments then come from ds_load_b128.

// x * splat(s): <16 x half> fmul with splat operand -> 8x v_pk_mul_f16
__device__ __forceinline__ v16h scale16(v16h x, _Float16 s) {
    v16h sv;
#pragma unroll
    for (int i = 0; i < 16; ++i) sv[i] = s;
    return x * sv;
}

__device__ __forceinline__ void issue_chunk(const _Float16* wt, int c, char* dst,
                                            unsigned tid) {
    uint64_t base = (uint64_t)(uintptr_t)wt + (uint64_t)c * CHUNK_B;
#pragma unroll
    for (int j = 0; j < CHUNK_B / 256 / 16; ++j) {   // 8 x b128 per thread
        uint32_t off = (j * 256 + tid) * 16;
        uint32_t la  = (uint32_t)(uintptr_t)(dst + off);
        asm volatile("global_load_async_to_lds_b128 %0, %1, %2"
                     :: "v"(la), "v"(off), "s"(base) : "memory");
    }
}

__global__ __launch_bounds__(256) void cin_main(
    const _Float16* __restrict__ X,
    const _Float16* __restrict__ Wt0,
    const _Float16* __restrict__ Wt1,
    const _Float16* __restrict__ Wt2,
    const float* __restrict__ bias0,
    const float* __restrict__ bias1,
    const float* __restrict__ bias2,
    float* __restrict__ out)
{
    extern __shared__ char smem[];
    _Float16* xl = (_Float16*)(smem + OFF_XL);   // [h][row][batch]
    _Float16* xb = (_Float16*)(smem + OFF_XB);   // [m][row][batch]

    const unsigned tid = threadIdx.x;
    const int lane = tid & 31;
    const int wave = tid >> 5;
    const int row  = lane & 15;
    const int hsel = lane >> 4;
    const int bgrp = wave >> 2;            // 0,1 -> batches bgrp*4..+3
    const int gp   = wave & 3;             // N-group pair {2gp, 2gp+1}
    const int b0i  = blockIdx.x * 8;       // first batch of this block

    // --- persistent x0 A-fragments for this wave's 4 batches ---
    v16h xf[4];
#pragma unroll
    for (int bi = 0; bi < 4; ++bi) {
        const _Float16* xrow =
            X + (size_t)((b0i + bgrp * 4 + bi) * EMB + row) * NFEAT + hsel * 16;
        v8h xa = *(const v8h*)(xrow);
        v8h xbv = *(const v8h*)(xrow + 8);
#pragma unroll
        for (int i = 0; i < 8; ++i) { xf[bi][i] = xa[i]; xf[bi][8 + i] = xbv[i]; }
    }

    // --- stage layer-0 scalars: xb[(m*16+row)*8 + batch] for the 8 block batches ---
#pragma unroll
    for (int j = 0; j < 16; ++j) {
        int i  = j * 256 + tid;            // 4096 entries
        int bb = i & 7;
        int mr = i >> 3;
        int rr = mr & 15;
        int mm = mr >> 4;
        xb[i] = X[(size_t)((b0i + bb) * EMB + rr) * NFEAT + mm];
    }
    __syncthreads();

    const _Float16* wts[3]    = {Wt0, Wt1, Wt2};
    const float*    biases[3] = {bias0, bias1, bias2};
    const int       nhs[3]    = {32, 128, 128};

    const int g0 = gp * 2, g1 = gp * 2 + 1;
    const int boff0 = (g0 * 16 + row) * 32 + hsel * 16;   // B frag offset in a slab
    const int boff1 = (g1 * 16 + row) * 32 + hsel * 16;

#pragma unroll
    for (int layer = 0; layer < 3; ++layer) {
        const _Float16* wt      = wts[layer];
        const _Float16* scalbuf = (layer == 0) ? xb : xl;
        const int nchunks = nhs[layer] / CH;

        v8f C[4][2];
#pragma unroll
        for (int bi = 0; bi < 4; ++bi) {
            v8f z = {};
            C[bi][0] = z; C[bi][1] = z;
        }

        issue_chunk(wt, 0, smem, tid);

        for (int c = 0; c < nchunks; ++c) {
            const _Float16* cur = (const _Float16*)(smem + (c & 1) * CHUNK_B);
            if (c + 1 < nchunks) {
                issue_chunk(wt, c + 1, smem + ((c + 1) & 1) * CHUNK_B, tid);
                asm volatile("s_wait_asynccnt 0x8" ::: "memory");
            } else {
                asm volatile("s_wait_asynccnt 0x0" ::: "memory");
            }
            __syncthreads();   // chunk c resident; safe to read

#pragma unroll
            for (int hl = 0; hl < CH; ++hl) {
                const int hg = c * CH + hl;
                // 4 per-batch scalars in one ds_load_b64
                v4h s4 = *(const v4h*)(scalbuf + (hg * 16 + row) * 8 + bgrp * 4);
                const _Float16* sl = cur + hl * (SLAB_B / 2);
                const v16h bf0 = *(const v16h*)(sl + boff0);
                const v16h bf1 = *(const v16h*)(sl + boff1);

                // build ALL four A fragments first (distinct overlapping live
                // ranges -> distinct registers; WMMA WAR hazards covered by
                // useful work).  scale16 keeps the packed-f16 mul form.
                v16h af0 = scale16(xf[0], s4[0]);
                v16h af1 = scale16(xf[1], s4[1]);
                v16h af2 = scale16(xf[2], s4[2]);
                v16h af3 = scale16(xf[3], s4[3]);

                C[0][0] = __builtin_amdgcn_wmma_f32_16x16x32_f16(
                    false, af0, false, bf0, (short)0, C[0][0], false, false);
                C[0][1] = __builtin_amdgcn_wmma_f32_16x16x32_f16(
                    false, af0, false, bf1, (short)0, C[0][1], false, false);
                C[1][0] = __builtin_amdgcn_wmma_f32_16x16x32_f16(
                    false, af1, false, bf0, (short)0, C[1][0], false, false);
                C[1][1] = __builtin_amdgcn_wmma_f32_16x16x32_f16(
                    false, af1, false, bf1, (short)0, C[1][1], false, false);
                C[2][0] = __builtin_amdgcn_wmma_f32_16x16x32_f16(
                    false, af2, false, bf0, (short)0, C[2][0], false, false);
                C[2][1] = __builtin_amdgcn_wmma_f32_16x16x32_f16(
                    false, af2, false, bf1, (short)0, C[2][1], false, false);
                C[3][0] = __builtin_amdgcn_wmma_f32_16x16x32_f16(
                    false, af3, false, bf0, (short)0, C[3][0], false, false);
                C[3][1] = __builtin_amdgcn_wmma_f32_16x16x32_f16(
                    false, af3, false, bf1, (short)0, C[3][1], false, false);
            }
            __syncthreads();   // all waves done with chunk c before it is overwritten
        }

        // ---- epilogue: +bias, relu (layers 0,1), write xl, column-sum output ----
#pragma unroll
        for (int bi = 0; bi < 4; ++bi) {
            const int bb = b0i + bgrp * 4 + bi;
#pragma unroll
            for (int g = 0; g < 2; ++g) {
                const int gg  = gp * 2 + g;
                const int col = gg * 16 + row;
                float bias_v = biases[layer][col];
                float psum = 0.f;
#pragma unroll
                for (int v = 0; v < 8; ++v) {
                    float y = C[bi][g][v] + bias_v;
                    if (layer < 2) y = fmaxf(y, 0.f);
                    psum += y;
                    if (layer < 2)
                        xl[(col * 16 + hsel * 8 + v) * 8 + (bgrp * 4 + bi)] =
                            (_Float16)y;
                }
                psum += __shfl_xor(psum, 16, 32);
                if (hsel == 0)
                    out[(size_t)bb * 384 + layer * HID + col] = psum;
            }
        }
        __syncthreads();       // xl complete before next layer reads it
    }
}

// ---------------- launcher ----------------

extern "C" void kernel_launch(void* const* d_in, const int* in_sizes, int n_in,
                              void* d_out, int out_size, void* d_ws, size_t ws_size,
                              hipStream_t stream) {
    const float* inputs = (const float*)d_in[0];
    const float* W0 = (const float*)d_in[1];
    const float* b0 = (const float*)d_in[2];
    const float* W1 = (const float*)d_in[3];
    const float* b1 = (const float*)d_in[4];
    const float* W2 = (const float*)d_in[5];
    const float* b2 = (const float*)d_in[6];

    _Float16* ws  = (_Float16*)d_ws;
    _Float16* X   = ws;                   // 524288 halves
    _Float16* Wt0 = X   + 524288;         // 131072 halves
    _Float16* Wt1 = Wt0 + 131072;         // 524288 halves
    _Float16* Wt2 = Wt1 + 524288;         // 524288 halves

    prep_x<<<(BATCH * EMB * NFEAT + 255) / 256, 256, 0, stream>>>(inputs, X);
    prep_w<<<(32  * HID * NFEAT + 255) / 256, 256, 0, stream>>>(W0, Wt0, 32);
    prep_w<<<(128 * HID * NFEAT + 255) / 256, 256, 0, stream>>>(W1, Wt1, 128);
    prep_w<<<(128 * HID * NFEAT + 255) / 256, 256, 0, stream>>>(W2, Wt2, 128);

    cin_main<<<BATCH / 8, 256, LDS_BYTES, stream>>>(X, Wt0, Wt1, Wt2,
                                                    b0, b1, b2, (float*)d_out);
}
